// DeformableAttention_89550068122287
// MI455X (gfx1250) — matrix operand order repaired and checked by reading, not compiled, not run on hardware
//
#include <hip/hip_runtime.h>
#include <hip/hip_bf16.h>
#include <math.h>

// Problem constants (match reference)
#define C_   256
#define NH_  8
#define NP_  4
#define B_   8
#define H_   96
#define W_   96
#define P_   (H_*W_)      // 9216 pixels per image
#define EPS_ 1e-5f

typedef __bf16 bf16;
typedef __attribute__((ext_vector_type(16))) __bf16 v16bf;
typedef __attribute__((ext_vector_type(8)))  __bf16 v8bf;
typedef __attribute__((ext_vector_type(8)))  float  v8f;

union ABfrag { v8bf h[2]; v16bf v; };

// ---------------------------------------------------------------------------
// Kernel 0: convert the four 1x1-conv weight matrices fp32 -> bf16 (row-major
// (O,K), K contiguous — exactly what the WMMA fragment loads want).
// ---------------------------------------------------------------------------
__global__ void k_prep_weights(const float* __restrict__ wv, const float* __restrict__ wo,
                               const float* __restrict__ wof, const float* __restrict__ wat,
                               bf16* __restrict__ dwv, bf16* __restrict__ dwo,
                               bf16* __restrict__ dwof, bf16* __restrict__ dwat) {
  int i = blockIdx.x * 256 + threadIdx.x;
  if (i < 65536) { dwv[i] = (bf16)wv[i]; dwo[i] = (bf16)wo[i]; }
  if (i < 16384) dwof[i] = (bf16)wof[i];
  if (i < 8192)  dwat[i] = (bf16)wat[i];
}

// ---------------------------------------------------------------------------
// Kernel 1: fused dwconv3x3 + BN + ReLU for BOTH branches, plus x -> bf16
// cast, with an LDS 32x32 tile transpose: coalesced NCHW reads, coalesced
// pixel-major (B,P,C) bf16 writes (channel contiguous for the GEMMs).
// ---------------------------------------------------------------------------
__global__ void k_dw_bn_relu_t(const float* __restrict__ x,
    const float* __restrict__ owdw, const float* __restrict__ obdw,
    const float* __restrict__ obg, const float* __restrict__ obb,
    const float* __restrict__ obm, const float* __restrict__ obv,
    const float* __restrict__ awdw, const float* __restrict__ abdw,
    const float* __restrict__ abg, const float* __restrict__ abb,
    const float* __restrict__ abm, const float* __restrict__ abv,
    bf16* __restrict__ xT, bf16* __restrict__ tof, bf16* __restrict__ tat) {
  __shared__ bf16 sx[32][34], so[32][34], sa[32][34]; // pad 2 -> conflict-free
  const int b  = blockIdx.z;
  const int c0 = blockIdx.y * 32;
  const int p0 = blockIdx.x * 32;
  const int tx = threadIdx.x, ty = threadIdx.y;

  #pragma unroll
  for (int sub = 0; sub < 4; ++sub) {
    const int cl = ty + 8 * sub;
    const int c  = c0 + cl;
    const int p  = p0 + tx;
    const int py = p / W_, px = p % W_;
    const float* xb = x + ((size_t)b * C_ + c) * P_;
    float accO = 0.f, accA = 0.f;
    #pragma unroll
    for (int dy = -1; dy <= 1; ++dy) {
      const int yy = py + dy;
      if (yy < 0 || yy >= H_) continue;
      #pragma unroll
      for (int dx = -1; dx <= 1; ++dx) {
        const int xx = px + dx;
        if (xx < 0 || xx >= W_) continue;
        const float v = xb[yy * W_ + xx];
        const int wi = c * 9 + (dy + 1) * 3 + (dx + 1);
        accO = fmaf(owdw[wi], v, accO);
        accA = fmaf(awdw[wi], v, accA);
      }
    }
    accO += obdw[c];  accA += abdw[c];
    const float scO = obg[c] * rsqrtf(obv[c] + EPS_);
    const float scA = abg[c] * rsqrtf(abv[c] + EPS_);
    accO = (accO - obm[c]) * scO + obb[c];
    accA = (accA - abm[c]) * scA + abb[c];
    accO = accO > 0.f ? accO : 0.f;
    accA = accA > 0.f ? accA : 0.f;
    sx[cl][tx] = (bf16)xb[p];
    so[cl][tx] = (bf16)accO;
    sa[cl][tx] = (bf16)accA;
  }
  __syncthreads();
  #pragma unroll
  for (int sub = 0; sub < 4; ++sub) {
    const int pl = ty + 8 * sub;
    const size_t o = ((size_t)b * P_ + p0 + pl) * C_ + c0 + tx; // channel contiguous
    xT[o]  = sx[tx][pl];
    tof[o] = so[tx][pl];
    tat[o] = sa[tx][pl];
  }
}

// ---------------------------------------------------------------------------
// Kernel 2: pixel-major GEMM  Out[b,p,o] = sum_k Wm[o,k]*T[b,p,k] + bias[o].
// o-blocked: each wave computes NO 16-row o-subtiles x 16 pixels, sharing ONE
// activation B fragment across NO WMMAs per k-step (4x less activation
// re-streaming vs per-16-o tiling; ~13 flop/byte on the load stream).
// A = 16x32 weight tile, B = 32x16 pixel tile; K=256 -> 8 k-steps, 8*NO WMMAs.
// D layout: lane holds column pixel n=lane&15, rows o -> epilogue stores 8
// contiguous channels per lane (pixel-major output).
// MODE: 0 = f32, 1 = f32+sigmoid, 2 = bf16.
// ---------------------------------------------------------------------------
template <int MODE, int OSTRIDE, int NO>
__global__ void k_gemm_pm(const bf16* __restrict__ Wm, const bf16* __restrict__ T,
                          const float* __restrict__ bias, void* __restrict__ outv) {
  const int lane = threadIdx.x & 31;
  const int wv   = threadIdx.x >> 5;
  const int b    = blockIdx.z;
  const int o0   = blockIdx.y * (16 * NO);
  const int p0   = (blockIdx.x * 4 + wv) * 16;
  const bf16* tb = T + (size_t)b * P_ * C_;

  // A (16x32 bf16): lane -> row m=lane&15, 16 contiguous K at (lane>>4)*16
  const int m  = lane & 15;
  const int kb = (lane >> 4) * 16;
  const bf16* arow = Wm + (size_t)(o0 + m) * C_ + kb;
  // B (32x16 bf16): lane -> col n=lane&15, 16 contiguous K at (lane>>4)*16
  const int n  = lane & 15;
  const int kc = (lane >> 4) * 16;
  const bf16* bcol = tb + (size_t)(p0 + n) * C_ + kc;

  v8f acc[NO];
  #pragma unroll
  for (int t = 0; t < NO; ++t) acc[t] = (v8f){};

  #pragma unroll
  for (int ks = 0; ks < C_; ks += 32) {
    const v16bf bfv = *(const v16bf*)(bcol + ks);        // shared across NO tiles
    #pragma unroll
    for (int t = 0; t < NO; ++t) {
      ABfrag a;
      a.h[0] = *(const v8bf*)(arow + (size_t)t * 16 * C_ + ks);
      a.h[1] = *(const v8bf*)(arow + (size_t)t * 16 * C_ + ks + 8);
      acc[t] = __builtin_amdgcn_wmma_f32_16x16x32_bf16(false, a.v, false, bfv,
                                                       (short)0, acc[t], false, false);
    }
  }

  const int p = p0 + n;
  #pragma unroll
  for (int t = 0; t < NO; ++t) {
    const int orow = o0 + t * 16 + (lane >> 4) * 8;
    const v8f bv = *(const v8f*)(bias + orow);
    const size_t outbase = ((size_t)b * P_ + p) * OSTRIDE + orow;
    if (MODE == 2) {
      v8bf r;
      #pragma unroll
      for (int j = 0; j < 8; ++j) r[j] = (bf16)(acc[t][j] + bv[j]);
      *(v8bf*)((bf16*)outv + outbase) = r;
    } else {
      v8f r;
      #pragma unroll
      for (int j = 0; j < 8; ++j) {
        float v = acc[t][j] + bv[j];
        if (MODE == 1) v = 1.0f / (1.0f + __expf(-v));
        r[j] = v;
      }
      *(v8f*)((float*)outv + outbase) = r;
    }
  }
}

// ---------------------------------------------------------------------------
// Kernel 3: deformable bilinear sampling + attention-weighted sum.
// One wave per (b, head, pixel); lane = channel within head (dh=32).
// value/agg are pixel-major bf16 -> each corner gather is a coalesced 64B
// wave read; agg write is coalesced.
// ix = 0.5 + px + offx*(W-1)/2  (exact algebraic reduction of the reference
// align_corners grid: ref_x = (0.5+i)/(W-1)*2-1, ix=(gx+1)*0.5*(W-1)).
// ---------------------------------------------------------------------------
__global__ void k_sample(const float* __restrict__ offs, const float* __restrict__ attn,
                         const bf16* __restrict__ val, bf16* __restrict__ agg) {
  const int lane = threadIdx.x & 31;
  const int wv   = threadIdx.x >> 5;
  const int g    = blockIdx.x * 8 + wv;
  const int p    = g % P_;
  const int rem  = g / P_;
  const int h    = rem & 7;
  const int b    = rem >> 3;

  const float* op = offs + ((size_t)b * P_ + p) * 64 + h * 8; // (pt,2) pairs
  const float* ap = attn + ((size_t)b * P_ + p) * 32 + h * 4;
  const bf16*  vb = val + (size_t)b * P_ * C_ + h * 32 + lane;
  const int py = p / W_, px = p % W_;

  auto fetch = [&](int xx, int yy) -> float {
    if (xx < 0 || xx >= W_ || yy < 0 || yy >= H_) return 0.f;
    return (float)vb[(size_t)(yy * W_ + xx) * C_];
  };

  float acc = 0.f;
  #pragma unroll
  for (int pt = 0; pt < 4; ++pt) {
    const float ix = 0.5f + px + op[pt * 2 + 0] * 47.5f;
    const float iy = 0.5f + py + op[pt * 2 + 1] * 47.5f;
    const float xf = floorf(ix), yf = floorf(iy);
    const int   x0 = (int)xf,    y0 = (int)yf;
    const float wx1 = ix - xf, wx0 = 1.f - wx1;
    const float wy1 = iy - yf, wy0 = 1.f - wy1;
    const float a = ap[pt];
    float s = wx0 * wy0 * fetch(x0,     y0)
            + wx1 * wy0 * fetch(x0 + 1, y0)
            + wx0 * wy1 * fetch(x0,     y0 + 1)
            + wx1 * wy1 * fetch(x0 + 1, y0 + 1);
    acc = fmaf(a, s, acc);
  }
  agg[((size_t)b * P_ + p) * C_ + h * 32 + lane] = (bf16)acc;
}

// ---------------------------------------------------------------------------
// Kernel 4: final 1x1 conv + bias + residual -> NCHW f32 output.
// Swapped WMMA orientation: A = 16-pixel x 32-K tile of agg (loaded ONCE per
// k-step), B = NO weight subtiles -> agg is streamed only O/(16*NO) times.
// D rows are PIXELS -> each lane stores 8 contiguous f32 in NCHW (and reads
// the residual x with the same contiguous access).
// ---------------------------------------------------------------------------
template <int NO>
__global__ void k_gemm_out(const bf16* __restrict__ agg, const bf16* __restrict__ Wo,
                           const float* __restrict__ bias, const float* __restrict__ x,
                           float* __restrict__ out) {
  const int lane = threadIdx.x & 31;
  const int wv   = threadIdx.x >> 5;
  const int b    = blockIdx.z;
  const int o0   = blockIdx.y * (16 * NO);
  const int p0   = (blockIdx.x * 4 + wv) * 16;

  const int m  = lane & 15;           // pixel row of A
  const int kb = (lane >> 4) * 16;
  const bf16* arow = agg + ((size_t)b * P_ + p0 + m) * C_ + kb;
  const int n  = lane & 15;           // output-channel col of B
  const int kc = (lane >> 4) * 16;
  const bf16* bcol = Wo + (size_t)(o0 + n) * C_ + kc;

  v8f acc[NO];
  #pragma unroll
  for (int t = 0; t < NO; ++t) acc[t] = (v8f){};

  #pragma unroll
  for (int ks = 0; ks < C_; ks += 32) {
    ABfrag a;                                            // shared across NO tiles
    a.h[0] = *(const v8bf*)(arow + ks);
    a.h[1] = *(const v8bf*)(arow + ks + 8);
    #pragma unroll
    for (int t = 0; t < NO; ++t) {
      const v16bf bfv = *(const v16bf*)(bcol + (size_t)t * 16 * C_ + ks);
      acc[t] = __builtin_amdgcn_wmma_f32_16x16x32_bf16(false, a.v, false, bfv,
                                                       (short)0, acc[t], false, false);
    }
  }

  const int prow = p0 + (lane >> 4) * 8;
  #pragma unroll
  for (int t = 0; t < NO; ++t) {
    const int o = o0 + t * 16 + n;
    const float bo = bias[o];
    const size_t base = ((size_t)b * C_ + o) * P_ + prow;  // NCHW, pixel contiguous
    const v8f r = *(const v8f*)(x + base);
    v8f d;
    #pragma unroll
    for (int j = 0; j < 8; ++j) d[j] = acc[t][j] + bo + r[j];
    *(v8f*)(out + base) = d;
  }
}

// ---------------------------------------------------------------------------
extern "C" void kernel_launch(void* const* d_in, const int* in_sizes, int n_in,
                              void* d_out, int out_size, void* d_ws, size_t ws_size,
                              hipStream_t stream) {
  const float* x        = (const float*)d_in[0];
  const float* off_dw_w = (const float*)d_in[1];
  const float* off_dw_b = (const float*)d_in[2];
  const float* off_bn_g = (const float*)d_in[3];
  const float* off_bn_b = (const float*)d_in[4];
  const float* off_bn_m = (const float*)d_in[5];
  const float* off_bn_v = (const float*)d_in[6];
  const float* off_pw_w = (const float*)d_in[7];
  const float* off_pw_b = (const float*)d_in[8];
  const float* att_dw_w = (const float*)d_in[9];
  const float* att_dw_b = (const float*)d_in[10];
  const float* att_bn_g = (const float*)d_in[11];
  const float* att_bn_b = (const float*)d_in[12];
  const float* att_bn_m = (const float*)d_in[13];
  const float* att_bn_v = (const float*)d_in[14];
  const float* att_pw_w = (const float*)d_in[15];
  const float* att_pw_b = (const float*)d_in[16];
  const float* val_w    = (const float*)d_in[17];
  const float* val_b    = (const float*)d_in[18];
  const float* out_w    = (const float*)d_in[19];
  const float* out_b    = (const float*)d_in[20];

  // Workspace layout (all big arrays bf16 pixel-major (B,P,C); agg aliases tof
  // which is fully consumed by the offset GEMM before sampling runs).
  char* ws = (char*)d_ws;
  bf16* wv_b  = (bf16*)(ws + 0);        // 256*256 bf16
  bf16* wo_b  = (bf16*)(ws + 131072);   // 256*256 bf16
  bf16* wof_b = (bf16*)(ws + 262144);   // 64*256  bf16
  bf16* wat_b = (bf16*)(ws + 294912);   // 32*256  bf16
  const size_t BIG = (size_t)B_ * P_ * C_ * sizeof(bf16);   // 37,748,736
  char* big0 = ws + (1u << 20);
  bf16*  xT   = (bf16*)(big0);
  bf16*  tof  = (bf16*)(big0 + 1 * BIG);
  bf16*  tat  = (bf16*)(big0 + 2 * BIG);
  bf16*  val  = (bf16*)(big0 + 3 * BIG);
  float* offs = (float*)(big0 + 4 * BIG);                   // B*P*64 f32
  float* attn = (float*)(big0 + 4 * BIG + (size_t)B_ * P_ * 64 * 4); // B*P*32 f32
  bf16*  agg  = tof;  // reuse

  // 0) weights -> bf16
  k_prep_weights<<<256, 256, 0, stream>>>(val_w, out_w, off_pw_w, att_pw_w,
                                          wv_b, wo_b, wof_b, wat_b);
  // 1) fused dwconv+BN+ReLU (both branches) + x cast, transposed to pixel-major
  dim3 g1(P_ / 32, C_ / 32, B_);  // 288 x 8 x 8
  k_dw_bn_relu_t<<<g1, dim3(32, 8), 0, stream>>>(x,
      off_dw_w, off_dw_b, off_bn_g, off_bn_b, off_bn_m, off_bn_v,
      att_dw_w, att_dw_b, att_bn_g, att_bn_b, att_bn_m, att_bn_v,
      xT, tof, tat);
  // 2) value projection (O=256, NO=4 -> grid.y = 4, bf16 out)
  dim3 gv(P_ / 64, 4, B_);
  k_gemm_pm<2, 256, 4><<<gv, 128, 0, stream>>>(wv_b, xT, val_b, (void*)val);
  // 3) offset conv (O=64, NO=4 -> grid.y = 1, f32 out)
  dim3 go(P_ / 64, 1, B_);
  k_gemm_pm<0, 64, 4><<<go, 128, 0, stream>>>(wof_b, tof, off_pw_b, (void*)offs);
  // 4) attention conv (O=32, NO=2 -> grid.y = 1, sigmoid, f32 out)
  dim3 ga(P_ / 64, 1, B_);
  k_gemm_pm<1, 32, 2><<<ga, 128, 0, stream>>>(wat_b, tat, att_pw_b, (void*)attn);
  // 5) bilinear sampling + attention-weighted sum over points
  k_sample<<<(B_ * NH_ * P_) / 8, 256, 0, stream>>>(offs, attn, val, agg);
  // 6) output conv + bias + residual -> NCHW f32 (NO=4 -> grid.y = 4)
  dim3 gf(P_ / 64, 4, B_);
  k_gemm_out<4><<<gf, 128, 0, stream>>>(agg, wo_b, out_b, x, (float*)d_out);
}
